// MultiHeadedCrossAttention_88794153877526
// MI455X (gfx1250) — compile-verified, matrix operand order checked by reading, not run on hardware
//
#include <hip/hip_runtime.h>
#include <hip/hip_bf16.h>

// ---------------------------------------------------------------------------
// MHA forward for MI455X (gfx1250, wave32, WMMA, TDM).
//   1) X -> bf16 ; W{q,k,v,h} -> transposed bf16 (Wt[n][k])
//   2) Q = (X@Wq)/8, K = X@Wk, V = X@Wv   (TDM double-buffered bf16 WMMA GEMM)
//   3) flash attention, 64-key tiles; K tiles TDM-streamed into LDS (double
//      buffered), V tiles cooperatively transposed (double buffered), row-sums
//      of P computed by a WMMA ones-matmul.
//   4) out = Hb@Wh + bh (fp32 epilogue)
// ---------------------------------------------------------------------------

typedef unsigned short u16;
typedef __attribute__((ext_vector_type(16))) __bf16 v16bf;
typedef __attribute__((ext_vector_type(8)))  float  v8f;
typedef __attribute__((ext_vector_type(4)))  unsigned int u32x4;
typedef __attribute__((ext_vector_type(8)))  int i32x8;
typedef __attribute__((ext_vector_type(4)))  int i32x4;

#define D_MODEL 1024
#define NHEAD   16
#define DK      64
#define BATCH   4
#define SEQ     2048
#define MTOT    (BATCH * SEQ)   // 8192 rows
#define LDSK    40              // padded K-stride (halfs) for GEMM LDS tiles

#if __has_builtin(__builtin_amdgcn_tensor_load_to_lds) && \
    __has_builtin(__builtin_amdgcn_s_wait_tensorcnt)
#define USE_TDM 1
#else
#define USE_TDM 0
#endif

union FragBF { v16bf v; uint4 q[2]; };

__device__ __forceinline__ u16 f2bf(float f) {
  unsigned u = __float_as_uint(f);
  u += 0x7FFFu + ((u >> 16) & 1u);     // round-to-nearest-even
  return (u16)(u >> 16);
}

// bf16 WMMA fragment load: lane's row base + koff (=8 for lanes 16..31).
// Chunks [koff..koff+7] and [koff+16..koff+23] (A 16x32 / Bt 32x16 layout).
__device__ __forceinline__ v16bf load_frag(const u16* base, int koff) {
  FragBF f;
  f.q[0] = *(const uint4*)(base + koff);
  f.q[1] = *(const uint4*)(base + koff + 16);
  return f.v;
}

#if USE_TDM
// Issue one TDM 2D tile load: `rows` rows x `tile0` bf16 elems from a [*, ldk]
// bf16 matrix into LDS at lds_byte, LDS-padded by 4 DWORDs (8 halfs) after
// every row (pad_interval code must match tile0: 2^(code+1) DWORDs == tile0/2).
// Wave-level op, tracked by TENSORcnt.
__device__ __forceinline__ void tdm_load_tile(const u16* g, unsigned lds_byte,
                                              int ldk, int tile0, int rows,
                                              int padIntCode) {
  unsigned long long ga = (unsigned long long)(size_t)g;
  u32x4 g0;
  g0[0] = 1u;                                   // count=1 valid user descriptor
  g0[1] = lds_byte;                             // LDS byte address
  g0[2] = (unsigned)ga;                         // global addr [31:0]
  g0[3] = (unsigned)((ga >> 32) & 0x01FFFFFFu) | 0x80000000u; // [56:32]|type=2
  i32x8 g1;
  g1[0] = (1 << 16) | (1 << 20) | (padIntCode << 22) | (3 << 25);
                                                // 2B elems, pad_en, pad=4dw
  g1[1] = (ldk & 0xFFFF) << 16;                 // tensor_dim0 lo16 (abar=0)
  g1[2] = ((unsigned)ldk >> 16) & 0xFFFF;       // tensor_dim0 hi16, dim1 lo16=0
  g1[3] = 0x10 | (tile0 << 16);                 // tensor_dim1=1M rows, tile_dim0
  g1[4] = rows;                                 // tile_dim1, tile_dim2=0
  g1[5] = ldk;                                  // tensor_dim0_stride lo32
  g1[6] = 0;
  g1[7] = 0;
  i32x4 z4 = {0, 0, 0, 0};
#if defined(__clang_major__) && (__clang_major__ >= 23)
  i32x8 z8 = {0, 0, 0, 0, 0, 0, 0, 0};
  __builtin_amdgcn_tensor_load_to_lds(g0, g1, z4, z4, z8, 0);
#else
  __builtin_amdgcn_tensor_load_to_lds(g0, g1, z4, z4, 0);
#endif
}
#endif

// ------------------------------- converts ----------------------------------

__global__ __launch_bounds__(256) void cvt_bf16_kernel(const float* __restrict__ in,
                                                       u16* __restrict__ out, int n4) {
  int i = blockIdx.x * 256 + threadIdx.x;
  if (i < n4) {
    float4 v = ((const float4*)in)[i];
    ushort4 o;
    o.x = f2bf(v.x); o.y = f2bf(v.y); o.z = f2bf(v.z); o.w = f2bf(v.w);
    ((ushort4*)out)[i] = o;
  }
}

__global__ __launch_bounds__(256) void cvt_transpose_kernel(const float* __restrict__ in,
                                                            u16* __restrict__ out,
                                                            int K, int N) {
  int idx = blockIdx.x * 256 + threadIdx.x;
  if (idx < N * K) {
    int n = idx / K, k = idx - n * K;
    out[idx] = f2bf(in[(size_t)k * N + n]);
  }
}

// ------------------------------- GEMM --------------------------------------
// C[M,N] = alpha * A[M,K](bf16) * Bt[N,K](bf16)^T (+bias, fp32 out if OUTF32)
// 256 thr (8 waves), 128x128 tile, BK=32, wave tile 64x32.
// TDM double-buffers the LDS tiles; all 8 waves compute while tile t+1 streams.
template <int OUTF32>
__global__ __launch_bounds__(256)
void gemm_bf16_kernel(const u16* __restrict__ A, const u16* __restrict__ Bt,
                      u16* __restrict__ Cb, float* __restrict__ Cf,
                      const float* __restrict__ bias, float alpha,
                      int M, int N, int K) {
  __shared__ u16 As[2][128 * LDSK];
  __shared__ u16 Bs[2][128 * LDSK];

  const int tid  = threadIdx.x;
  const int wid  = tid >> 5;
  const int lane = tid & 31;
  const int l16  = lane & 15;
  const int half = lane >> 4;
  const int koff = half * 8;
  const int wm   = wid >> 2;
  const int wn   = wid & 3;
  const int blockM = blockIdx.y * 128;
  const int blockN = blockIdx.x * 128;
  const int nTiles = K >> 5;

  v8f acc[4][2] = {};

#if USE_TDM
  const int wid_u = __builtin_amdgcn_readfirstlane(wid);  // uniform -> scalar branch
  const unsigned ldsA = (unsigned)(size_t)(void*)&As[0][0];
  const unsigned ldsB = (unsigned)(size_t)(void*)&Bs[0][0];
  const unsigned bufBytes = 128 * LDSK * 2;               // 10240 B per buffer
  if (wid_u == 0) {
    tdm_load_tile(A + (size_t)blockM * K, ldsA, K, 32, 128, 3);
    tdm_load_tile(Bt + (size_t)blockN * K, ldsB, K, 32, 128, 3);
  }
#endif

  for (int t = 0; t < nTiles; ++t) {
#if USE_TDM
    const int cb = t & 1;
    if (wid_u == 0) {
      if (t + 1 < nTiles) {   // stream next tile into the other buffer
        tdm_load_tile(A + (size_t)blockM * K + (t + 1) * 32,
                      ldsA + (unsigned)(cb ^ 1) * bufBytes, K, 32, 128, 3);
        tdm_load_tile(Bt + (size_t)blockN * K + (t + 1) * 32,
                      ldsB + (unsigned)(cb ^ 1) * bufBytes, K, 32, 128, 3);
        __builtin_amdgcn_s_wait_tensorcnt(2);   // current tile's pair done
      } else {
        __builtin_amdgcn_s_wait_tensorcnt(0);
      }
    }
    __syncthreads();
#else
    const int cb = 0;
    const int k0 = t << 5;
#pragma unroll
    for (int i = 0; i < 2; ++i) {
      int ch  = tid + i * 256;
      int row = ch >> 2;
      int kc  = (ch & 3) << 3;
      const u16* ga = A  + (size_t)(blockM + row) * K + k0 + kc;
      const u16* gb = Bt + (size_t)(blockN + row) * K + k0 + kc;
      *(uint4*)&As[0][row * LDSK + kc] = *(const uint4*)ga;
      *(uint4*)&Bs[0][row * LDSK + kc] = *(const uint4*)gb;
      if (k0 + 32 < K) {
        __builtin_prefetch(ga + 32, 0, 0);
        __builtin_prefetch(gb + 32, 0, 0);
      }
    }
    __syncthreads();
#endif

    v16bf b0 = load_frag(&Bs[cb][(wn * 32 + l16) * LDSK], koff);
    v16bf b1 = load_frag(&Bs[cb][(wn * 32 + 16 + l16) * LDSK], koff);
#pragma unroll
    for (int mt = 0; mt < 4; ++mt) {
      v16bf a = load_frag(&As[cb][(wm * 64 + mt * 16 + l16) * LDSK], koff);
      acc[mt][0] = __builtin_amdgcn_wmma_f32_16x16x32_bf16(
          false, a, false, b0, (short)0, acc[mt][0], false, false);
      acc[mt][1] = __builtin_amdgcn_wmma_f32_16x16x32_bf16(
          false, a, false, b1, (short)0, acc[mt][1], false, false);
    }
    __syncthreads();
  }

#pragma unroll
  for (int mt = 0; mt < 4; ++mt) {
#pragma unroll
    for (int nt = 0; nt < 2; ++nt) {
      int col = blockN + wn * 32 + nt * 16 + l16;
      int r0  = blockM + wm * 64 + mt * 16 + half * 8;
#pragma unroll
      for (int i = 0; i < 8; ++i) {
        float v = acc[mt][nt][i] * alpha;
        size_t off = (size_t)(r0 + i) * N + col;
        if (OUTF32) Cf[off] = v + bias[col];
        else        Cb[off] = f2bf(v);
      }
    }
  }
}

// --------------------------- flash attention -------------------------------
// Grid (SEQ/64, NHEAD, BATCH), 128 thr = 4 waves; each wave owns 16 q rows.
// 64-key tiles. K tile t+1 streams via TDM into LDS while tile t computes;
// V tile t+1 is cooperatively transposed into its double buffer at the top of
// the iteration so its global latency is covered by the score phase.
#define VSTR 72   // K/V LDS stride (64 + 8 pad halfs)
#define PSTR 72
#define KVBUF (DK * VSTR)   // halfs per K or V buffer

__global__ __launch_bounds__(128)
void attn_kernel(const u16* __restrict__ Qb, const u16* __restrict__ Kb,
                 const u16* __restrict__ Vb, u16* __restrict__ Hb) {
  __shared__ u16 Ks[2][KVBUF];            // K tile: [key64][d64] (TDM, padded)
  __shared__ u16 Vt[2][KVBUF];            // V^T tile: [d64][key64]
  __shared__ u16 Pt[4 * 16 * PSTR];       // per-wave P bounce: [16 q][64 key]

  const int tid  = threadIdx.x;
  const int wid  = tid >> 5;
  const int lane = tid & 31;
  const int l16  = lane & 15;
  const int half = lane >> 4;
  const int koff = half * 8;
  const int b    = blockIdx.z;
  const int h    = blockIdx.y;
  const int q0   = blockIdx.x * 64 + wid * 16;
  const size_t hcol = (size_t)h * DK;
  const int wid_u = __builtin_amdgcn_readfirstlane(wid);

  const u16* qrow = Qb + ((size_t)b * SEQ + q0 + l16) * D_MODEL + hcol;
  v16bf qf0 = load_frag(qrow, koff);        // d 0..31 (pre-scaled by 1/8)
  v16bf qf1 = load_frag(qrow + 32, koff);   // d 32..63

  FragBF ones;                              // bf16 1.0 = 0x3F80
  { uint4 o; o.x = o.y = o.z = o.w = 0x3F803F80u; ones.q[0] = o; ones.q[1] = o; }

  v8f hacc[4] = {};
  v8f lacc = {};
  float m[8];
#pragma unroll
  for (int i = 0; i < 8; ++i) m[i] = -3.0e38f;

  u16* pw = Pt + wid * 16 * PSTR;
  const u16* kbase = Kb + (size_t)b * SEQ * D_MODEL + hcol;
  const int key  = tid & 63;
  const int dseg = (tid >> 6) * 32;
  const int T = SEQ / 64;

  // ---- prologue: tile 0 into buffer 0 ----
#if USE_TDM
  const unsigned ldsK = (unsigned)(size_t)(void*)&Ks[0][0];
  if (wid_u == 0)
    tdm_load_tile(kbase, ldsK, D_MODEL, 64, 64, 4);
#endif
  {
    const u16* vp = Vb + ((size_t)b * SEQ + key) * D_MODEL + hcol + dseg;
    union { uint4 q[4]; u16 s[32]; } tmp;
    tmp.q[0] = *(const uint4*)vp;       tmp.q[1] = *(const uint4*)(vp + 8);
    tmp.q[2] = *(const uint4*)(vp + 16); tmp.q[3] = *(const uint4*)(vp + 24);
#pragma unroll
    for (int j = 0; j < 32; ++j) Vt[0][(dseg + j) * VSTR + key] = tmp.s[j];
#if !USE_TDM
    union { uint4 q[4]; u16 s[32]; } ktmp;
    const u16* kp = kbase + (size_t)key * D_MODEL + dseg;
    ktmp.q[0] = *(const uint4*)kp;       ktmp.q[1] = *(const uint4*)(kp + 8);
    ktmp.q[2] = *(const uint4*)(kp + 16); ktmp.q[3] = *(const uint4*)(kp + 24);
#pragma unroll
    for (int j = 0; j < 32; ++j) Ks[0][key * VSTR + dseg + j] = ktmp.s[j];
#endif
  }
#if USE_TDM
  if (wid_u == 0) __builtin_amdgcn_s_wait_tensorcnt(0);
#endif
  __syncthreads();

  for (int t = 0; t < T; ++t) {
    const int cb = t & 1;
    const int t0 = t * 64;

    // ---- start streaming tile t+1 into the other buffers ----
    if (t + 1 < T) {
#if USE_TDM
      if (wid_u == 0)
        tdm_load_tile(kbase + (size_t)(t0 + 64) * D_MODEL,
                      ldsK + (unsigned)(cb ^ 1) * (KVBUF * 2), D_MODEL, 64, 64, 4);
#endif
      const u16* vp = Vb + ((size_t)b * SEQ + t0 + 64 + key) * D_MODEL + hcol + dseg;
      union { uint4 q[4]; u16 s[32]; } tmp;
      tmp.q[0] = *(const uint4*)vp;       tmp.q[1] = *(const uint4*)(vp + 8);
      tmp.q[2] = *(const uint4*)(vp + 16); tmp.q[3] = *(const uint4*)(vp + 24);
#pragma unroll
      for (int j = 0; j < 32; ++j) Vt[cb ^ 1][(dseg + j) * VSTR + key] = tmp.s[j];
#if !USE_TDM
      union { uint4 q[4]; u16 s[32]; } ktmp;
      const u16* kp = kbase + (size_t)(t0 + 64 + key) * D_MODEL + dseg;
      ktmp.q[0] = *(const uint4*)kp;       ktmp.q[1] = *(const uint4*)(kp + 8);
      ktmp.q[2] = *(const uint4*)(kp + 16); ktmp.q[3] = *(const uint4*)(kp + 24);
#pragma unroll
      for (int j = 0; j < 32; ++j) Ks[cb ^ 1][key * VSTR + dseg + j] = ktmp.s[j];
#endif
    }

    // ---- scores S(16x64) = Q(16x64) . K^T, K frags from LDS ----
    v8f s[4] = {};
#pragma unroll
    for (int c = 0; c < 2; ++c) {
      v16bf qf = c ? qf1 : qf0;
#pragma unroll
      for (int nt = 0; nt < 4; ++nt) {
        v16bf kf = load_frag(&Ks[cb][(nt * 16 + l16) * VSTR + c * 32], koff);
        s[nt] = __builtin_amdgcn_wmma_f32_16x16x32_bf16(
            false, qf, false, kf, (short)0, s[nt], false, false);
      }
    }

    // ---- online softmax: row max (shfl butterfly over 16-lane halves) ----
    float mloc[8];
#pragma unroll
    for (int i = 0; i < 8; ++i)
      mloc[i] = fmaxf(fmaxf(s[0][i], s[1][i]), fmaxf(s[2][i], s[3][i]));
#pragma unroll
    for (int mask = 1; mask <= 8; mask <<= 1)
#pragma unroll
      for (int i = 0; i < 8; ++i)
        mloc[i] = fmaxf(mloc[i], __shfl_xor(mloc[i], mask, 32));

    float scale[8];
#pragma unroll
    for (int i = 0; i < 8; ++i) {
      float mn = fmaxf(m[i], mloc[i]);
      scale[i] = __expf(m[i] - mn);
      m[i] = mn;
    }
#pragma unroll
    for (int nt = 0; nt < 4; ++nt)
#pragma unroll
      for (int i = 0; i < 8; ++i)
        s[nt][i] = __expf(s[nt][i] - m[i]);    // s becomes P
#pragma unroll
    for (int i = 0; i < 8; ++i) lacc[i] *= scale[i];
#pragma unroll
    for (int nt = 0; nt < 4; ++nt)
#pragma unroll
      for (int i = 0; i < 8; ++i) hacc[nt][i] *= scale[i];

    // ---- P: C-layout -> A-layout via per-wave LDS bounce ----
#pragma unroll
    for (int nt = 0; nt < 4; ++nt)
#pragma unroll
      for (int i = 0; i < 8; ++i)
        pw[(i + half * 8) * PSTR + nt * 16 + l16] = f2bf(s[nt][i]);
    asm volatile("s_wait_dscnt 0" ::: "memory");
    v16bf pf0 = load_frag(pw + l16 * PSTR, koff);        // keys 0..31
    v16bf pf1 = load_frag(pw + l16 * PSTR + 32, koff);   // keys 32..63

    // ---- row sums by WMMA: l += P . ones  (lands in C-layout of lacc) ----
    lacc = __builtin_amdgcn_wmma_f32_16x16x32_bf16(
        false, pf0, false, ones.v, (short)0, lacc, false, false);
    lacc = __builtin_amdgcn_wmma_f32_16x16x32_bf16(
        false, pf1, false, ones.v, (short)0, lacc, false, false);

    // ---- H(16x64) += P(16x64) . V(64x64) ----
#pragma unroll
    for (int c = 0; c < 2; ++c) {
      v16bf pf = c ? pf1 : pf0;
#pragma unroll
      for (int nt = 0; nt < 4; ++nt) {
        v16bf vf = load_frag(&Vt[cb][(nt * 16 + l16) * VSTR + c * 32], koff);
        hacc[nt] = __builtin_amdgcn_wmma_f32_16x16x32_bf16(
            false, pf, false, vf, (short)0, hacc[nt], false, false);
      }
    }

#if USE_TDM
    if (wid_u == 0) __builtin_amdgcn_s_wait_tensorcnt(0);  // K(t+1) landed
#endif
    __syncthreads();   // K/V staging visible; buffers[cb] free for reuse
  }

  // ---- normalize and emit bf16 into concatenated-head layout ----
#pragma unroll
  for (int nt = 0; nt < 4; ++nt)
#pragma unroll
    for (int i = 0; i < 8; ++i) {
      int r = q0 + i + half * 8;
      Hb[((size_t)b * SEQ + r) * D_MODEL + hcol + nt * 16 + l16] =
          f2bf(hacc[nt][i] / lacc[i]);
    }
}

// ------------------------------ launcher -----------------------------------

extern "C" void kernel_launch(void* const* d_in, const int* in_sizes, int n_in,
                              void* d_out, int out_size, void* d_ws, size_t ws_size,
                              hipStream_t stream) {
  (void)in_sizes; (void)n_in; (void)out_size; (void)ws_size;
  const float* X  = (const float*)d_in[0];
  const float* Wq = (const float*)d_in[1];
  const float* Wk = (const float*)d_in[2];
  const float* Wv = (const float*)d_in[3];
  const float* Wh = (const float*)d_in[4];
  const float* bh = (const float*)d_in[5];
  float* out = (float*)d_out;

  const size_t XE = (size_t)MTOT * D_MODEL;
  const size_t WE = (size_t)D_MODEL * D_MODEL;
  u16* Xb  = (u16*)d_ws;
  u16* Wqt = Xb  + XE;
  u16* Wkt = Wqt + WE;
  u16* Wvt = Wkt + WE;
  u16* Wht = Wvt + WE;
  u16* Qb  = Wht + WE;
  u16* Kb  = Qb  + XE;
  u16* Vb  = Kb  + XE;
  u16* Hb  = Vb  + XE;   // ~92 MB of u16 total

  cvt_bf16_kernel<<<(int)(XE / 4 / 256), 256, 0, stream>>>(X, Xb, (int)(XE / 4));
  int tgrid = (int)((WE + 255) / 256);
  cvt_transpose_kernel<<<tgrid, 256, 0, stream>>>(Wq, Wqt, D_MODEL, D_MODEL);
  cvt_transpose_kernel<<<tgrid, 256, 0, stream>>>(Wk, Wkt, D_MODEL, D_MODEL);
  cvt_transpose_kernel<<<tgrid, 256, 0, stream>>>(Wv, Wvt, D_MODEL, D_MODEL);
  cvt_transpose_kernel<<<tgrid, 256, 0, stream>>>(Wh, Wht, D_MODEL, D_MODEL);

  dim3 gproj(D_MODEL / 128, MTOT / 128);
  gemm_bf16_kernel<0><<<gproj, 256, 0, stream>>>(Xb, Wqt, Qb, nullptr, nullptr,
                                                 0.125f, MTOT, D_MODEL, D_MODEL);
  gemm_bf16_kernel<0><<<gproj, 256, 0, stream>>>(Xb, Wkt, Kb, nullptr, nullptr,
                                                 1.0f, MTOT, D_MODEL, D_MODEL);
  gemm_bf16_kernel<0><<<gproj, 256, 0, stream>>>(Xb, Wvt, Vb, nullptr, nullptr,
                                                 1.0f, MTOT, D_MODEL, D_MODEL);

  dim3 gattn(SEQ / 64, NHEAD, BATCH);
  attn_kernel<<<gattn, 128, 0, stream>>>(Qb, Kb, Vb, Hb);

  gemm_bf16_kernel<1><<<gproj, 256, 0, stream>>>(Hb, Wht, nullptr, out, bh,
                                                 1.0f, MTOT, D_MODEL, D_MODEL);
}
